// M_head_t_59794534695352
// MI455X (gfx1250) — compile-verified, matrix operand order checked by reading, not run on hardware
//
#include <hip/hip_runtime.h>
#include <hip/hip_bf16.h>
#include <cstdint>

// ---------------------------------------------------------------------------
// M_head_t temporal attention for MI455X (gfx1250, wave32, WMMA).
//   B=4, T=256, N=256, D_IN=D_H=128.
//   Pipeline: 3x (X @ W + b) projections in f16 WMMA (f32 accumulate),
//   then per-head causal flash attention in f16 WMMA with async (DMA-style)
//   K/V tile staging into LDS, double-buffered via ASYNCcnt.
// ---------------------------------------------------------------------------

typedef __attribute__((ext_vector_type(16))) _Float16 v16h;
typedef __attribute__((ext_vector_type(8)))  _Float16 v8h;
typedef __attribute__((ext_vector_type(8)))  float    v8f;
typedef __attribute__((ext_vector_type(4)))  float    v4f;

#define BDIM 4
#define TDIM 256
#define NDIM 256
#define DDIM 128
#define NHEADS (BDIM * NDIM)          // 1024
#define NTOKENS (BDIM * TDIM * NDIM)  // 262144

// LDS row pitches (halves) padded to break 64-bank aliasing on ds_load_b128
#define KPITCH 136   // 128 + 8 halves  -> 272 B row (17 x 16 B)
#define VPITCH 40    // 32 + 8 halves   -> 80 B row  (5 x 16 B)

// ---------------------------------------------------------------------------
// Fragment loaders. Per the CDNA5 ISA 16-bit A/B layout for 16x16x32:
//   lane L (m = L&15, hi = L>>4) holds, for each 32-wide K chunk,
//   halves [K = 8*hi .. 8*hi+7] then [K = 16+8*hi .. 23+8*hi].
// With row-major f16 storage this is exactly two contiguous 16-byte loads.
// ---------------------------------------------------------------------------
__device__ __forceinline__ v16h frag_from_f16(const _Float16* row, int hi) {
  v8h lo = *(const v8h*)(row + hi * 8);
  v8h hh = *(const v8h*)(row + 16 + hi * 8);
  v16h r;
#pragma unroll
  for (int i = 0; i < 8; ++i) { r[i] = lo[i]; r[i + 8] = hh[i]; }
  return r;
}

__device__ __forceinline__ v16h frag_from_f32(const float* row, int hi) {
  const v4f* p = (const v4f*)(row + hi * 8);
  const v4f* q = (const v4f*)(row + 16 + hi * 8);
  v4f a0 = p[0], a1 = p[1], b0 = q[0], b1 = q[1];
  v16h r;
#pragma unroll
  for (int i = 0; i < 4; ++i) {
    r[i]      = (_Float16)a0[i];
    r[i + 4]  = (_Float16)a1[i];
    r[i + 8]  = (_Float16)b0[i];
    r[i + 12] = (_Float16)b1[i];
  }
  return r;
}

__device__ __forceinline__ v8f wmma_f16(v16h a, v16h b, v8f c) {
  return __builtin_amdgcn_wmma_f32_16x16x32_f16(
      /*neg_a=*/false, a, /*neg_b=*/false, b,
      /*c_mod=*/(short)0, c, /*reuse_a=*/false, /*reuse_b=*/false);
}

// ---------------------------------------------------------------------------
// Async global -> LDS copy (GLOBAL_LOAD_ASYNC_TO_LDS_B128, ASYNCcnt-tracked).
// Inline asm for toolchain portability. LDS address = low 32 bits of the
// generic pointer (ISA: flat->LDS aperture truncates addr to LDS_ADDR[31:0]).
// ---------------------------------------------------------------------------
__device__ __forceinline__ void async_copy_b128(_Float16* lds_dst,
                                                const _Float16* gsrc) {
  uint32_t loff = (uint32_t)(uintptr_t)lds_dst;
  asm volatile("global_load_async_to_lds_b128 %0, %1, off"
               :: "v"(loff), "v"(gsrc) : "memory");
}

// ---------------------------------------------------------------------------
// Kernel 1: Y = X @ W + b, X fp32 [NTOKENS,128], W fp32 [128,128].
// mode 0/1: write f16 Y as [head][t][128]   (Q, K)
// mode 2  : write f16 Y as [head][128][T]   (V transposed, for PV WMMA)
// Block = 256 threads = 8 waves; each wave owns a 16-token M tile, all 8
// 16-wide N tiles, K=128 in 4 chunks -> 32 WMMAs per wave.
// ---------------------------------------------------------------------------
__global__ __launch_bounds__(256) void qkv_proj_kernel(
    const float* __restrict__ x, const float* __restrict__ W,
    const float* __restrict__ bias, _Float16* __restrict__ out, int mode) {
  __shared__ _Float16 wt[DDIM * DDIM];  // Wt[n][k] = W[k][n], 32 KB

  int tid = threadIdx.x;
  for (int idx = tid; idx < DDIM * DDIM; idx += 256) {
    int n = idx >> 7, k = idx & 127;
    wt[idx] = (_Float16)W[k * DDIM + n];
  }
  __syncthreads();

  int lane = tid & 31;
  int wave = tid >> 5;
  int hi = lane >> 4;
  int ln = lane & 15;

  long token0 = ((long)blockIdx.x * 8 + wave) * 16;
  const float* xrow = x + (token0 + ln) * DDIM;

  v16h A[4];
#pragma unroll
  for (int c = 0; c < 4; ++c) A[c] = frag_from_f32(xrow + 32 * c, hi);

#pragma unroll
  for (int j = 0; j < 8; ++j) {
    int col = j * 16 + ln;
    v8f acc = {};
#pragma unroll
    for (int c = 0; c < 4; ++c) {
      v16h Bf = frag_from_f16(&wt[col * DDIM + 32 * c], hi);
      acc = wmma_f16(A[c], Bf, acc);
    }
    float bv = bias[col];
#pragma unroll
    for (int r = 0; r < 8; ++r) {
      long token = token0 + r + 8 * hi;       // C/D layout: row = r + 8*hi
      int node = (int)(token & (NDIM - 1));
      long bt = token >> 8;
      int t = (int)(bt & (TDIM - 1));
      int b = (int)(bt >> 8);
      long head = (long)b * NDIM + node;
      float v = acc[r] + bv;
      long addr;
      if (mode == 2) addr = (head * DDIM + col) * TDIM + t;   // V^T
      else           addr = (head * TDIM + t) * DDIM + col;   // Q, K
      out[addr] = (_Float16)v;
    }
  }
}

// ---------------------------------------------------------------------------
// Kernel 2: causal flash attention per head. Grid = NHEADS * 2 blocks; each
// block covers 128 query rows (one wave per 16 rows) of ONE head, so K/V
// tiles are shared: they are staged cooperatively into padded LDS buffers
// with async global->LDS copies, double-buffered on ASYNCcnt.
// ---------------------------------------------------------------------------
__global__ __launch_bounds__(256) void attn_kernel(
    const _Float16* __restrict__ Q, const _Float16* __restrict__ K,
    const _Float16* __restrict__ Vt, float* __restrict__ out) {
  __shared__ _Float16 kbuf[2][32 * KPITCH];   // 2 x 8.5 KB
  __shared__ _Float16 vbuf[2][128 * VPITCH];  // 2 x 10 KB
  __shared__ _Float16 pbuf[8][16 * 32];       // per-wave P staging, 8 KB

  int tid = threadIdx.x;
  int lane = tid & 31, wave = tid >> 5;
  int hi = lane >> 4, ln = lane & 15;

  int head = blockIdx.x >> 1;
  int qblk = blockIdx.x & 1;
  int q0 = qblk * 128 + wave * 16;
  int b = head >> 8, node = head & (NDIM - 1);

  const _Float16* Qh = Q + (long)head * TDIM * DDIM;
  const _Float16* Kh = K + (long)head * TDIM * DDIM;
  const _Float16* Vh = Vt + (long)head * DDIM * TDIM;

  // ---- stage helper: 4 async B128 issues per thread per 32-wide KV tile ---
  auto stage_tile = [&](int bufi, int s0) {
    _Float16* kb = kbuf[bufi];
    _Float16* vb = vbuf[bufi];
#pragma unroll
    for (int r = 0; r < 2; ++r) {           // K tile: 32 rows x 256 B
      int g = tid + r * 256;                // 512 x 16 B chunks
      int row = g >> 4, c = g & 15;
      async_copy_b128(kb + row * KPITCH + c * 8,
                      Kh + (long)s0 * DDIM + g * 8);
    }
#pragma unroll
    for (int r = 0; r < 2; ++r) {           // V^T slice: 128 rows x 64 B
      int g = tid + r * 256;                // 512 x 16 B chunks
      int row = g >> 2, c = g & 3;
      async_copy_b128(vb + row * VPITCH + c * 8,
                      Vh + (long)row * TDIM + s0 + c * 8);
    }
  };

  // ---- per-wave Q fragments (16 rows x 128) ------------------------------
  v16h Aq[4];
  const _Float16* qrow = Qh + (q0 + ln) * DDIM;
#pragma unroll
  for (int c = 0; c < 4; ++c) Aq[c] = frag_from_f16(qrow + 32 * c, hi);

  v8f zero = {};
  v8f acc[8];
#pragma unroll
  for (int j = 0; j < 8; ++j) acc[j] = zero;

  float row_max[8], row_sum[8];
#pragma unroll
  for (int r = 0; r < 8; ++r) { row_max[r] = -1e30f; row_sum[r] = 0.0f; }

  const float scale = 0.08838834764831845f;  // 1/sqrt(128)
  const int s_end = q0 + 15;                 // this wave's causal bound
  const int ntiles = (qblk + 1) * 4;         // block-wide KV tile count

  stage_tile(0, 0);                          // prologue DMA

  for (int it = 0; it < ntiles; ++it) {
    int s0 = it * 32;
    int cur = it & 1;
    bool pref = (it + 1 < ntiles);
    if (pref) stage_tile(cur ^ 1, s0 + 32);  // overlap next tile's DMA
    // Async loads retire in order: ASYNCcnt<=4 means current tile landed
    // while the 4 just-issued prefetch copies stay in flight.
    if (pref) asm volatile("s_wait_asynccnt 4" ::: "memory");
    else      asm volatile("s_wait_asynccnt 0" ::: "memory");
    __syncthreads();

    if (s0 <= s_end) {  // wave-level causal skip (barriers stay outside)
      const _Float16* kb = kbuf[cur];
      const _Float16* vb = vbuf[cur];

      // ---- S = Q K^T for 32 KV columns (from LDS) ----------------------
      const _Float16* krow0 = kb + ln * KPITCH;
      const _Float16* krow1 = kb + (16 + ln) * KPITCH;
      v8f S0 = {}, S1 = {};
#pragma unroll
      for (int c = 0; c < 4; ++c) {
        v16h B0 = frag_from_f16(krow0 + 32 * c, hi);
        S0 = wmma_f16(Aq[c], B0, S0);
        v16h B1 = frag_from_f16(krow1 + 32 * c, hi);
        S1 = wmma_f16(Aq[c], B1, S1);
      }

      // ---- online softmax (rows live across one 16-lane half) ----------
      float pcorr[8];
#pragma unroll
      for (int r = 0; r < 8; ++r) {
        int trow = q0 + r + 8 * hi;
        float a0 = S0[r] * scale; if (s0 + ln > trow)      a0 = -1e9f;
        float a1 = S1[r] * scale; if (s0 + 16 + ln > trow) a1 = -1e9f;
        float rm = fmaxf(a0, a1);
#pragma unroll
        for (int m = 1; m < 16; m <<= 1) rm = fmaxf(rm, __shfl_xor(rm, m, 32));
        float mn = fmaxf(row_max[r], rm);
        float p0 = __expf(a0 - mn);
        float p1 = __expf(a1 - mn);
        float ps = p0 + p1;
#pragma unroll
        for (int m = 1; m < 16; m <<= 1) ps += __shfl_xor(ps, m, 32);
        float corr = __expf(row_max[r] - mn);
        row_sum[r] = row_sum[r] * corr + ps;
        row_max[r] = mn;
        pcorr[r] = corr;
        int rr = r + 8 * hi;
        pbuf[wave][rr * 32 + ln]      = (_Float16)p0;
        pbuf[wave][rr * 32 + 16 + ln] = (_Float16)p1;
      }
#pragma unroll
      for (int j = 0; j < 8; ++j)
#pragma unroll
        for (int r = 0; r < 8; ++r) acc[j][r] *= pcorr[r];

      // wave-private LDS round trip (f32 D-layout -> f16 A-layout)
      asm volatile("s_wait_dscnt 0" ::: "memory");
      v16h P = frag_from_f16(&pbuf[wave][ln * 32], hi);

      // ---- O += P @ V (V^T rows contiguous along s, from LDS) ----------
#pragma unroll
      for (int j = 0; j < 8; ++j) {
        const _Float16* vrow = vb + (j * 16 + ln) * VPITCH;
        v16h Bv = frag_from_f16(vrow, hi);
        acc[j] = wmma_f16(P, Bv, acc[j]);
      }
    }
    __syncthreads();  // all waves done with buf[cur] before it is re-staged
  }

  // ---- epilogue: normalize and scatter to [B, T, N, Dh] fp32 -------------
#pragma unroll
  for (int r = 0; r < 8; ++r) {
    int t = q0 + r + 8 * hi;
    float inv = 1.0f / row_sum[r];
    long base = (((long)b * TDIM + t) * NDIM + node) * (long)DDIM;
#pragma unroll
    for (int j = 0; j < 8; ++j) {
      out[base + j * 16 + ln] = acc[j][r] * inv;
    }
  }
}

// ---------------------------------------------------------------------------
extern "C" void kernel_launch(void* const* d_in, const int* in_sizes, int n_in,
                              void* d_out, int out_size, void* d_ws, size_t ws_size,
                              hipStream_t stream) {
  const float* x  = (const float*)d_in[0];
  const float* Wq = (const float*)d_in[1];
  const float* bq = (const float*)d_in[2];
  const float* Wk = (const float*)d_in[3];
  const float* bk = (const float*)d_in[4];
  const float* Wv = (const float*)d_in[5];
  const float* bv = (const float*)d_in[6];
  float* out = (float*)d_out;

  const long TOT = (long)NTOKENS * DDIM;  // 33.55M f16 elements each
  _Float16* Qb = (_Float16*)d_ws;
  _Float16* Kb = Qb + TOT;
  _Float16* Vb = Kb + TOT;

  dim3 blk(256);
  int proj_blocks = NTOKENS / (16 * 8);   // 2048
  qkv_proj_kernel<<<proj_blocks, blk, 0, stream>>>(x, Wq, bq, Qb, 0);
  qkv_proj_kernel<<<proj_blocks, blk, 0, stream>>>(x, Wk, bk, Kb, 1);
  qkv_proj_kernel<<<proj_blocks, blk, 0, stream>>>(x, Wv, bv, Vb, 2);

  attn_kernel<<<NHEADS * 2, blk, 0, stream>>>(Qb, Kb, Vb, out);
}
